// MultiHeadAttention_27943057228343
// MI455X (gfx1250) — matrix operand order repaired and checked by reading, not compiled, not run on hardware
//
#include <hip/hip_runtime.h>
#include <hip/hip_bf16.h>

typedef __attribute__((ext_vector_type(16))) __bf16 v16bf;
typedef __attribute__((ext_vector_type(8)))  float  v8f;

#define WMMA_BF16(a, b, c) \
  __builtin_amdgcn_wmma_f32_16x16x32_bf16(false, (a), false, (b), (short)0, (c), false, false)

// ---------------------------------------------------------------------------
// Fragment loaders (NT layout: source row-major, K contiguous).
// A operand 16x32: lane l<16 -> row l, K 0..15;
//                  lane l>=16 -> row l-16, K 16..31.
// B operand 32x16 (stored as 16 rows of B^T, K contiguous):
//                  lane l<16 -> col l, K 0..15; lane l>=16 -> col l-16, K 16..31.
// ---------------------------------------------------------------------------
static __device__ __forceinline__ v16bf fragA_bf16(const __bf16* p, int ld) {
  const int l = threadIdx.x & 31;
  const __bf16* q = p + (l & 15) * ld + ((l & 16) ? 16 : 0);
  v16bf f;
#pragma unroll
  for (int i = 0; i < 16; ++i) f[i] = q[i];
  return f;
}

static __device__ __forceinline__ v16bf fragA_f32(const float* p, int ld) {
  const int l = threadIdx.x & 31;
  const float* q = p + (l & 15) * ld + ((l & 16) ? 16 : 0);
  v16bf f;
#pragma unroll
  for (int i = 0; i < 16; ++i) f[i] = (__bf16)q[i];
  return f;
}

static __device__ __forceinline__ v16bf fragB_bf16(const __bf16* p, int ld) {
  const int l = threadIdx.x & 31;
  const __bf16* q = p + (l & 15) * ld + ((l & 16) ? 16 : 0);
  v16bf f;
#pragma unroll
  for (int i = 0; i < 16; ++i) f[i] = q[i];
  return f;
}

static __device__ __forceinline__ v16bf fragB_f32(const float* p, int ld) {
  const int l = threadIdx.x & 31;
  const float* q = p + (l & 15) * ld + ((l & 16) ? 16 : 0);
  v16bf f;
#pragma unroll
  for (int i = 0; i < 16; ++i) f[i] = (__bf16)q[i];
  return f;
}

// LDS byte offset of a generic pointer known to be in LDS.
static __device__ __forceinline__ unsigned lds_offset(const void* p) {
  return (unsigned)(unsigned long long)(const __attribute__((address_space(3))) char*)p;
}

// Async 16-byte global->LDS copy (per lane), tracked by ASYNCcnt.
static __device__ __forceinline__ void async_ld16(unsigned lds_off, const void* gptr) {
  asm volatile("global_load_async_to_lds_b128 %0, %1, off"
               :: "v"(lds_off), "v"(gptr) : "memory");
}

// ---------------------------------------------------------------------------
// Kernel 1: projection  Y = X @ W^T + b, f32 in -> bf16 out.
// Wave computes a 64x32 macro-tile (4 M-subtiles x 2 N-subtiles): 8 WMMAs per
// 6 fragment loads per k-step. Block = 8 waves = 256x64 tile. Grid (16,16).
// mode 0: write (b,h,s,dk)   [Q, K];  mode 1: write (b,h,dk,s)  [V]
// ---------------------------------------------------------------------------
__global__ __launch_bounds__(256) void mha_proj_kernel(
    const float* __restrict__ X, const float* __restrict__ W,
    const float* __restrict__ bias, __bf16* __restrict__ dst, int mode)
{
  const int wave = threadIdx.x >> 5;
  const int lane = threadIdx.x & 31;
  const int wm = wave & 3;
  const int wn = wave >> 2;
  const int tm0 = blockIdx.y * 16 + wm * 4;
  const int tn0 = blockIdx.x * 4 + wn * 2;

  v8f acc[4][2] = {};
#pragma unroll 2
  for (int k = 0; k < 1024; k += 32) {
    v16bf a[4], b[2];
#pragma unroll
    for (int i = 0; i < 4; ++i)
      a[i] = fragA_f32(X + (size_t)(tm0 + i) * 16 * 1024 + k, 1024);
#pragma unroll
    for (int j = 0; j < 2; ++j)
      b[j] = fragB_f32(W + (size_t)(tn0 + j) * 16 * 1024 + k, 1024);
#pragma unroll
    for (int i = 0; i < 4; ++i)
#pragma unroll
      for (int j = 0; j < 2; ++j)
        acc[i][j] = WMMA_BF16(a[i], b[j], acc[i][j]);
  }

#pragma unroll
  for (int i = 0; i < 4; ++i) {
#pragma unroll
    for (int j = 0; j < 2; ++j) {
      const int n = (tn0 + j) * 16 + (lane & 15);
      const float bvv = bias[n];
      const int h = n >> 6, d = n & 63;
#pragma unroll
      for (int r = 0; r < 8; ++r) {
        const int m  = (tm0 + i) * 16 + r + ((lane & 16) ? 8 : 0);
        const int b_ = m >> 11, s = m & 2047;
        const float v = acc[i][j][r] + bvv;
        const size_t idx = (mode == 0)
            ? (((size_t)(b_ * 16 + h) * 2048 + s) * 64 + d)
            : (((size_t)(b_ * 16 + h) * 64 + d) * 2048 + s);
        dst[idx] = (__bf16)v;
      }
    }
  }
}

// ---------------------------------------------------------------------------
// Kernel 2: flash attention, double-buffered async LDS pipeline.
// Block = 8 waves sharing one (b,h), 8 consecutive q-tiles. 64-key chunks of
// K (64x64) and V (64x64) staged in LDS via global_load_async_to_lds_b128;
// fills for chunk i+1 overlap with compute on chunk i. One barrier per chunk.
// ---------------------------------------------------------------------------
__global__ __launch_bounds__(256) void mha_attn_kernel(
    const __bf16* __restrict__ Qh, const __bf16* __restrict__ Kh,
    const __bf16* __restrict__ Vt, __bf16* __restrict__ AO)
{
  __shared__ __bf16 smK[2][64 * 64];   // 2 x 8 KB: 64 keys x 64 dk
  __shared__ __bf16 smV[2][64 * 64];   // 2 x 8 KB: 64 d    x 64 keys
  __shared__ __bf16 smP[8][16 * 64];   // 16 KB: per-wave P tiles

  const int tid  = threadIdx.x;
  const int wave = tid >> 5;
  const int lane = tid & 31;
  const int bh = blockIdx.x >> 4;               // 0..31
  const int qt = (blockIdx.x & 15) * 8 + wave;  // 0..127

  const __bf16* Qp = Qh + ((size_t)bh * 2048 + qt * 16) * 64;
  const __bf16* Kp = Kh + (size_t)bh * 2048 * 64;
  const __bf16* Vp = Vt + (size_t)bh * 64 * 2048;
  __bf16* lp = &smP[wave][0];

  // Cooperative fill: 8KB tile = 512 x 16B slices; thread t does slices t, t+256.
  const int rowA = tid >> 3, seg = tid & 7;     // slice t   : row 0..31
  const int rowB = rowA + 32;                   // slice t+256: row 32..63
  const __bf16* kgA = Kp + (size_t)rowA * 64 + seg * 8;
  const __bf16* kgB = Kp + (size_t)rowB * 64 + seg * 8;
  const __bf16* vgA = Vp + (size_t)rowA * 2048 + seg * 8;
  const __bf16* vgB = Vp + (size_t)rowB * 2048 + seg * 8;
  unsigned koffA[2], koffB[2], voffA[2], voffB[2];
#pragma unroll
  for (int b = 0; b < 2; ++b) {
    koffA[b] = lds_offset(&smK[b][rowA * 64 + seg * 8]);
    koffB[b] = lds_offset(&smK[b][rowB * 64 + seg * 8]);
    voffA[b] = lds_offset(&smV[b][rowA * 64 + seg * 8]);
    voffB[b] = lds_offset(&smV[b][rowB * 64 + seg * 8]);
  }
  auto fill = [&](int b, int kt) {
    async_ld16(koffA[b], kgA + (size_t)kt * 64);
    async_ld16(koffB[b], kgB + (size_t)kt * 64);
    async_ld16(voffA[b], vgA + kt);
    async_ld16(voffB[b], vgB + kt);
  };

  const v16bf qa0 = fragA_bf16(Qp, 64);       // dk 0..31
  const v16bf qa1 = fragA_bf16(Qp + 32, 64);  // dk 32..63

  v8f o0 = {}, o1 = {}, o2 = {}, o3 = {};
  float mrun[8], lrun[8];
#pragma unroll
  for (int r = 0; r < 8; ++r) { mrun[r] = -1e30f; lrun[r] = 0.0f; }

  const float SC = 0.125f;  // 1/sqrt(64)
  fill(0, 0);               // prologue: stage chunk 0

  for (int it = 0; it < 32; ++it) {
    const int buf = it & 1;
    asm volatile("s_wait_asynccnt 0x0" ::: "memory");  // current chunk landed
    __syncthreads();  // everyone's slices visible; buf^1 free to overwrite
    if (it + 1 < 32) fill(buf ^ 1, (it + 1) * 64);     // overlap with compute
    const __bf16* sk = &smK[buf][0];
    const __bf16* sv = &smV[buf][0];

    // Scores: 4 tiles of 16x16 over the 64-key chunk.
    v8f s[4];
#pragma unroll
    for (int j = 0; j < 4; ++j) {
      v8f z = {};
      z = WMMA_BF16(qa0, fragB_bf16(sk + j * 16 * 64, 64), z);
      z = WMMA_BF16(qa1, fragB_bf16(sk + j * 16 * 64 + 32, 64), z);
      s[j] = z;
    }

#pragma unroll
    for (int r = 0; r < 8; ++r) {
      const float a0 = s[0][r] * SC, a1 = s[1][r] * SC;
      const float a2 = s[2][r] * SC, a3 = s[3][r] * SC;
      float mx = fmaxf(fmaxf(a0, a1), fmaxf(a2, a3));
#pragma unroll
      for (int off = 8; off > 0; off >>= 1)
        mx = fmaxf(mx, __shfl_xor(mx, off, 32));   // stays within 16-lane half
      const float mnew = fmaxf(mrun[r], mx);
      const float corr = __expf(mrun[r] - mnew);
      const float p0 = __expf(a0 - mnew);
      const float p1 = __expf(a1 - mnew);
      const float p2 = __expf(a2 - mnew);
      const float p3 = __expf(a3 - mnew);
      float ps = (p0 + p1) + (p2 + p3);
#pragma unroll
      for (int off = 8; off > 0; off >>= 1)
        ps += __shfl_xor(ps, off, 32);
      lrun[r] = lrun[r] * corr + ps;
      mrun[r] = mnew;
      o0[r] *= corr; o1[r] *= corr; o2[r] *= corr; o3[r] *= corr;
      const int row = r + ((lane & 16) ? 8 : 0);
      const int c = lane & 15;
      lp[row * 64 + c]      = (__bf16)p0;
      lp[row * 64 + 16 + c] = (__bf16)p1;
      lp[row * 64 + 32 + c] = (__bf16)p2;
      lp[row * 64 + 48 + c] = (__bf16)p3;
    }
    asm volatile("s_wait_dscnt 0x0" ::: "memory");  // P tile visible within wave
    const v16bf pf0 = fragA_bf16(lp, 64);        // keys 0..31
    const v16bf pf1 = fragA_bf16(lp + 32, 64);   // keys 32..63
    o0 = WMMA_BF16(pf0, fragB_bf16(sv + 0 * 16 * 64, 64), o0);
    o0 = WMMA_BF16(pf1, fragB_bf16(sv + 0 * 16 * 64 + 32, 64), o0);
    o1 = WMMA_BF16(pf0, fragB_bf16(sv + 1 * 16 * 64, 64), o1);
    o1 = WMMA_BF16(pf1, fragB_bf16(sv + 1 * 16 * 64 + 32, 64), o1);
    o2 = WMMA_BF16(pf0, fragB_bf16(sv + 2 * 16 * 64, 64), o2);
    o2 = WMMA_BF16(pf1, fragB_bf16(sv + 2 * 16 * 64 + 32, 64), o2);
    o3 = WMMA_BF16(pf0, fragB_bf16(sv + 3 * 16 * 64, 64), o3);
    o3 = WMMA_BF16(pf1, fragB_bf16(sv + 3 * 16 * 64 + 32, 64), o3);
  }

  const int b_ = bh >> 4, h = bh & 15;
#pragma unroll
  for (int r = 0; r < 8; ++r) {
    const int srow = qt * 16 + r + ((lane & 16) ? 8 : 0);
    const float inv = 1.0f / lrun[r];
    __bf16* op = AO + ((size_t)b_ * 2048 + srow) * 1024 + h * 64 + (lane & 15);
    op[0]  = (__bf16)(o0[r] * inv);
    op[16] = (__bf16)(o1[r] * inv);
    op[32] = (__bf16)(o2[r] * inv);
    op[48] = (__bf16)(o3[r] * inv);
  }
}

// ---------------------------------------------------------------------------
// Kernel 3: output projection  out = AO @ Wo^T + bo  (bf16 x f32 -> f32),
// same 64x32-per-wave register blocking as kernel 1.
// ---------------------------------------------------------------------------
__global__ __launch_bounds__(256) void mha_oproj_kernel(
    const __bf16* __restrict__ A, const float* __restrict__ W,
    const float* __restrict__ bias, float* __restrict__ dst)
{
  const int wave = threadIdx.x >> 5;
  const int lane = threadIdx.x & 31;
  const int wm = wave & 3;
  const int wn = wave >> 2;
  const int tm0 = blockIdx.y * 16 + wm * 4;
  const int tn0 = blockIdx.x * 4 + wn * 2;

  v8f acc[4][2] = {};
#pragma unroll 2
  for (int k = 0; k < 1024; k += 32) {
    v16bf a[4], b[2];
#pragma unroll
    for (int i = 0; i < 4; ++i)
      a[i] = fragA_bf16(A + (size_t)(tm0 + i) * 16 * 1024 + k, 1024);
#pragma unroll
    for (int j = 0; j < 2; ++j)
      b[j] = fragB_f32(W + (size_t)(tn0 + j) * 16 * 1024 + k, 1024);
#pragma unroll
    for (int i = 0; i < 4; ++i)
#pragma unroll
      for (int j = 0; j < 2; ++j)
        acc[i][j] = WMMA_BF16(a[i], b[j], acc[i][j]);
  }

#pragma unroll
  for (int i = 0; i < 4; ++i) {
#pragma unroll
    for (int j = 0; j < 2; ++j) {
      const int n = (tn0 + j) * 16 + (lane & 15);
      const float bvv = bias[n];
#pragma unroll
      for (int r = 0; r < 8; ++r) {
        const int m = (tm0 + i) * 16 + r + ((lane & 16) ? 8 : 0);
        dst[(size_t)m * 1024 + n] = acc[i][j][r] + bvv;
      }
    }
  }
}

// ---------------------------------------------------------------------------
extern "C" void kernel_launch(void* const* d_in, const int* in_sizes, int n_in,
                              void* d_out, int out_size, void* d_ws, size_t ws_size,
                              hipStream_t stream) {
  (void)in_sizes; (void)n_in; (void)out_size; (void)ws_size;
  const float* query = (const float*)d_in[0];
  const float* key_  = (const float*)d_in[1];
  const float* value = (const float*)d_in[2];
  const float* Wq = (const float*)d_in[3];
  const float* bq = (const float*)d_in[4];
  const float* Wk = (const float*)d_in[5];
  const float* bk = (const float*)d_in[6];
  const float* Wv = (const float*)d_in[7];
  const float* bv = (const float*)d_in[8];
  const float* Wo = (const float*)d_in[9];
  const float* bo = (const float*)d_in[10];
  float* out = (float*)d_out;

  const size_t HE = (size_t)2 * 16 * 2048 * 64;  // elements per head-split tensor
  __bf16* Qh = (__bf16*)d_ws;   // 8 MB (b,h,s,dk)
  __bf16* Kh = Qh + HE;         // 8 MB (b,h,s,dk)
  __bf16* Vt = Kh + HE;         // 8 MB (b,h,dk,s)
  __bf16* AO = Vt + HE;         // 8 MB (b,s,d_model)

  const dim3 pb(256), pg(16, 16);
  mha_proj_kernel<<<pg, pb, 0, stream>>>(query, Wq, bq, Qh, 0);
  mha_proj_kernel<<<pg, pb, 0, stream>>>(key_,  Wk, bk, Kh, 0);
  mha_proj_kernel<<<pg, pb, 0, stream>>>(value, Wv, bv, Vt, 1);
  mha_attn_kernel<<<dim3(512), pb, 0, stream>>>(Qh, Kh, Vt, AO);
  mha_oproj_kernel<<<pg, pb, 0, stream>>>(AO, Wo, bo, out);
}
